// GAT_15814069584089
// MI455X (gfx1250) — matrix operand order repaired and checked by reading, not compiled, not run on hardware
//
#include <hip/hip_runtime.h>
#include <hip/hip_bf16.h>

// ---------------------------------------------------------------------------
// GAT (3 layers, N=4096, F=256) for MI455X / gfx1250, wave32 + WMMA bf16,
// async global->LDS double-buffered GEMM pipeline.
// ---------------------------------------------------------------------------

typedef unsigned short u16;
typedef __attribute__((ext_vector_type(16))) __bf16 v16bf;
typedef __attribute__((ext_vector_type(8)))  float  v8f;

#define GAT_N    4096
#define GAT_F    256
#define ALPHA    0.01f
#define NEG_INF  (-9.0e15f)

// float -> bf16 round-to-nearest-even
__device__ __forceinline__ u16 f2bf(float x) {
  unsigned u = __builtin_bit_cast(unsigned, x);
  unsigned r = (u + 0x7FFFu + ((u >> 16) & 1u)) >> 16;
  return (u16)r;
}

// ---------------------------------------------------------------------------
// CDNA5 async global->LDS copy (16 bytes per lane), ASYNCcnt-tracked.
// Probed signature: (int4 addrspace(1)* gsrc, int4 addrspace(3)* ldst,
//                    imm offset, imm cpol)
// ---------------------------------------------------------------------------
__device__ __forceinline__ void async_copy16(const void* g, const void* lds_generic) {
  unsigned loff = (unsigned)(unsigned long long)(lds_generic);  // low 32 bits = LDS offset
#if __has_builtin(__builtin_amdgcn_global_load_async_to_lds_b128)
  typedef __attribute__((ext_vector_type(4))) int v4i;
  typedef __attribute__((address_space(1))) v4i* gp;
  typedef __attribute__((address_space(3))) v4i* lp;
  __builtin_amdgcn_global_load_async_to_lds_b128(
      (gp)(unsigned long long)g, (lp)loff, 0, 0);
#else
  asm volatile("global_load_async_to_lds_b128 %0, %1, off"
               :: "v"(loff), "v"((unsigned long long)g)
               : "memory");
#endif
}

__device__ __forceinline__ void wait_async0() {
#if __has_builtin(__builtin_amdgcn_s_wait_asynccnt)
  __builtin_amdgcn_s_wait_asynccnt(0);
#else
  asm volatile("s_wait_asynccnt 0x0" ::: "memory");
#endif
}

// ---------------------------------------------------------------------------
// elementwise f32 -> bf16 (row-major)
// ---------------------------------------------------------------------------
__global__ void cvt_bf16_kernel(const float* __restrict__ x, u16* __restrict__ y, int n) {
  int i = blockIdx.x * blockDim.x + threadIdx.x;
  if (i < n) y[i] = f2bf(x[i]);
}

// f32 [R,C] row-major -> bf16 transposed [C,R]  (for GEMM B operands)
__global__ void cvt_bf16_t_kernel(const float* __restrict__ x, u16* __restrict__ y,
                                  int R, int C) {
  int i = blockIdx.x * blockDim.x + threadIdx.x;
  if (i < R * C) {
    int r = i / C, c = i % C;
    y[(size_t)c * R + r] = f2bf(x[i]);
  }
}

// ---------------------------------------------------------------------------
// pack adj (int32 0/1, 4096x4096) into bitmask: 128 words per row
// ---------------------------------------------------------------------------
__global__ void pack_adj_kernel(const int* __restrict__ adj, unsigned* __restrict__ out) {
  int w = blockIdx.x * blockDim.x + threadIdx.x;
  const int4* p = (const int4*)(adj + (size_t)w * 32);
  unsigned bits = 0;
#pragma unroll
  for (int q = 0; q < 8; ++q) {
    int4 v = p[q];
    bits |= (unsigned)(v.x > 0) << (q * 4 + 0);
    bits |= (unsigned)(v.y > 0) << (q * 4 + 1);
    bits |= (unsigned)(v.z > 0) << (q * 4 + 2);
    bits |= (unsigned)(v.w > 0) << (q * 4 + 3);
  }
  out[w] = bits;
}

// ---------------------------------------------------------------------------
// Tiled bf16 WMMA GEMM (TN form):  C[M,N] = A[M,K] @ Bt[N,K]^T
// A and Bt both row-major with contiguous K -> identical async staging paths.
// Block: 256 threads (8 wave32s), tile 128(M) x 64(N), K step 64 (2 WMMA steps).
// Double-buffered LDS, async global->LDS copies, s_wait_asynccnt + 1 barrier/iter.
// Outputs (each optional): Cf f32 row-major, Cb bf16 row-major, Cbt bf16 transposed.
// act=1 applies ELU.
// ---------------------------------------------------------------------------
#define BM 128
#define BN 64
#define BK 64
#define BKP 72          // padded row stride (u16): 144B, 16B-aligned, spreads banks

union FragU { uint4 q[2]; v16bf v; };

__global__ __launch_bounds__(256)
void gemm_bf16_wmma(const u16* __restrict__ A, const u16* __restrict__ Bt,
                    float* Cf, u16* Cb, u16* Cbt,
                    int M, int N, int K, int act) {
  __shared__ __align__(16) u16 As[2][BM][BKP];
  __shared__ __align__(16) u16 Bs[2][BN][BKP];

  const int tid  = threadIdx.x;
  const int lane = tid & 31;
  const int wave = tid >> 5;
  const int m0 = blockIdx.y * BM;
  const int n0 = blockIdx.x * BN;
  const int wm = (wave >> 1) * 32;
  const int wn = (wave & 1) * 32;
  const int lr = lane & 15;
  const int hi = lane >> 4;

  const int nk = K / BK;

  // stage one 128x64 A tile + 64x64 Bt tile into LDS buffer `buf` (async)
  auto stage = [&](int buf, int k0) {
#pragma unroll
    for (int p = 0; p < 4; ++p) {               // A: 1024 16B chunks
      int c = tid + p * 256;
      int r = c >> 3, c8 = (c & 7) * 8;
      async_copy16(A + (size_t)(m0 + r) * K + k0 + c8, &As[buf][r][c8]);
    }
#pragma unroll
    for (int p = 0; p < 2; ++p) {               // Bt: 512 16B chunks
      int c = tid + p * 256;
      int r = c >> 3, c8 = (c & 7) * 8;
      async_copy16(Bt + (size_t)(n0 + r) * K + k0 + c8, &Bs[buf][r][c8]);
    }
  };

  v8f acc[2][2] = {};
  stage(0, 0);

  for (int kt = 0; kt < nk; ++kt) {
    wait_async0();          // this wave's async tile data is in LDS
    __syncthreads();        // everyone's is; also fences previous compute reads
    if (kt + 1 < nk) stage((kt + 1) & 1, (kt + 1) * BK);
    if (kt + 2 < nk) {      // L2 prefetch two tiles ahead
      __builtin_prefetch(A + (size_t)(m0 + (tid >> 1)) * K + (kt + 2) * BK, 0, 0);
      __builtin_prefetch(Bt + (size_t)(n0 + (tid >> 2)) * K + (kt + 2) * BK, 0, 0);
    }
    const int buf = kt & 1;
#pragma unroll
    for (int s = 0; s < 2; ++s) {               // two 16x16x32 K-steps per tile
      FragU a[2], b[2];
#pragma unroll
      for (int t = 0; t < 2; ++t) {
        const u16* ar = &As[buf][wm + t * 16 + lr][s * 32];
        a[t].q[0] = *(const uint4*)(ar + hi * 8);        // K: hi*8..+8
        a[t].q[1] = *(const uint4*)(ar + 16 + hi * 8);   // K: 16+hi*8..+8
        const u16* br = &Bs[buf][wn + t * 16 + lr][s * 32];
        b[t].q[0] = *(const uint4*)(br + hi * 8);        // K: hi*8..+8
        b[t].q[1] = *(const uint4*)(br + 16 + hi * 8);   // K: 16+hi*8..+8
      }
#pragma unroll
      for (int i = 0; i < 2; ++i)
#pragma unroll
        for (int j = 0; j < 2; ++j)
          acc[i][j] = __builtin_amdgcn_wmma_f32_16x16x32_bf16(
              false, a[i].v, false, b[j].v, (short)0, acc[i][j], false, false);
    }
  }

  // epilogue: C layout lane&15 = col, lane>>4 picks M-half, vgpr r = row
#pragma unroll
  for (int i = 0; i < 2; ++i) {
#pragma unroll
    for (int j = 0; j < 2; ++j) {
      const int col  = n0 + wn + j * 16 + lr;
      const int row0 = m0 + wm + i * 16 + hi * 8;
      float x[8];
#pragma unroll
      for (int r = 0; r < 8; ++r) {
        float v = acc[i][j][r];
        if (act) v = v > 0.f ? v : (__expf(v) - 1.f);    // ELU
        x[r] = v;
      }
      if (Cf) {
#pragma unroll
        for (int r = 0; r < 8; ++r) Cf[(size_t)(row0 + r) * N + col] = x[r];
      }
      if (Cb) {
#pragma unroll
        for (int r = 0; r < 8; ++r) Cb[(size_t)(row0 + r) * N + col] = f2bf(x[r]);
      }
      if (Cbt) {  // transposed bf16: lane's 8 rows are contiguous -> one b128 store
        union { u16 h[8]; uint4 q; } pk;
#pragma unroll
        for (int r = 0; r < 8; ++r) pk.h[r] = f2bf(x[r]);
        *(uint4*)(Cbt + (size_t)col * M + row0) = pk.q;
      }
    }
  }
}

// ---------------------------------------------------------------------------
// f1[i] = Wh[i,:] . a[0:256] ; f2[i] = Wh[i,:] . a[256:512]
// one wave32 per row, 8 rows per block
// ---------------------------------------------------------------------------
__global__ __launch_bounds__(256)
void gemv_f1f2(const float* __restrict__ Wh, const float* __restrict__ a,
               float* __restrict__ f1, float* __restrict__ f2) {
  int lane = threadIdx.x & 31, wave = threadIdx.x >> 5;
  int row = blockIdx.x * 8 + wave;
  const float* w = Wh + (size_t)row * GAT_F;
  float s1 = 0.f, s2 = 0.f;
#pragma unroll
  for (int t = 0; t < 8; ++t) {
    int k = lane + 32 * t;
    float x = w[k];
    s1 += x * a[k];
    s2 += x * a[GAT_F + k];
  }
#pragma unroll
  for (int o = 16; o; o >>= 1) {
    s1 += __shfl_xor(s1, o, 32);
    s2 += __shfl_xor(s2, o, 32);
  }
  if (lane == 0) { f1[row] = s1; f2[row] = s2; }
}

// ---------------------------------------------------------------------------
// Fused masked-leakyrelu-softmax per row -> bf16 attention matrix
// ---------------------------------------------------------------------------
__global__ __launch_bounds__(256)
void masked_softmax(const float* __restrict__ f1, const float* __restrict__ f2,
                    const unsigned* __restrict__ adjb, u16* __restrict__ att) {
  const int i = blockIdx.x, tid = threadIdx.x;
  const int lane = tid & 31, wave = tid >> 5;
  __shared__ float red[8];

  const float fi = f1[i];
  const unsigned* arow = adjb + (size_t)i * (GAT_N / 32);

  float v[16];
  float m = -3.4e38f;
#pragma unroll
  for (int t = 0; t < 16; ++t) {
    int j = tid + t * 256;
    float e = fi + f2[j];
    e = e > 0.f ? e : ALPHA * e;                       // leaky relu
    unsigned bit = (arow[j >> 5] >> (j & 31)) & 1u;
    v[t] = bit ? e : NEG_INF;
    m = fmaxf(m, v[t]);
  }
#pragma unroll
  for (int o = 16; o; o >>= 1) m = fmaxf(m, __shfl_xor(m, o, 32));
  if (lane == 0) red[wave] = m;
  __syncthreads();
  m = red[0];
#pragma unroll
  for (int w = 1; w < 8; ++w) m = fmaxf(m, red[w]);

  float p[16];
  float s = 0.f;
#pragma unroll
  for (int t = 0; t < 16; ++t) { p[t] = __expf(v[t] - m); s += p[t]; }
#pragma unroll
  for (int o = 16; o; o >>= 1) s += __shfl_xor(s, o, 32);
  __syncthreads();
  if (lane == 0) red[wave] = s;
  __syncthreads();
  s = red[0] + red[1] + red[2] + red[3] + red[4] + red[5] + red[6] + red[7];

  float inv = 1.f / s;
  u16* orow = att + (size_t)i * GAT_N;
#pragma unroll
  for (int t = 0; t < 16; ++t) orow[tid + t * 256] = f2bf(p[t] * inv);
}

// ---------------------------------------------------------------------------
// host-side orchestration
// ---------------------------------------------------------------------------
extern "C" void kernel_launch(void* const* d_in, const int* in_sizes, int n_in,
                              void* d_out, int out_size, void* d_ws, size_t ws_size,
                              hipStream_t stream) {
  const int Nn = GAT_N, F = GAT_F;
  const float* input = (const float*)d_in[0];
  const int*   adj   = (const int*)d_in[1];
  const float* W0 = (const float*)d_in[2];
  const float* Wl[3] = { (const float*)d_in[3], (const float*)d_in[5], (const float*)d_in[7] };
  const float* al[3] = { (const float*)d_in[4], (const float*)d_in[6], (const float*)d_in[8] };

  // workspace carve-up (~45 MB)
  char* base = (char*)d_ws;
  size_t off = 0;
  auto take = [&](size_t bytes) -> void* {
    void* p = base + off;
    off = (off + bytes + 255) & ~(size_t)255;
    return p;
  };
  u16*      att   = (u16*)take((size_t)Nn * Nn * 2);             // bf16 attention (A op)
  unsigned* adjb  = (unsigned*)take((size_t)Nn * (Nn / 32) * 4); // adjacency bitmask
  u16*      xB    = (u16*)take((size_t)Nn * F * 2);              // input bf16 (A op)
  u16*      hB    = (u16*)take((size_t)Nn * F * 2);              // h bf16 (A op)
  float*    WhF   = (float*)take((size_t)Nn * F * 4);            // Wh f32 (gemv)
  u16*      WhBt  = (u16*)take((size_t)Nn * F * 2);              // Wh^T bf16 [F][N] (B op)
  u16*      WtB[4];
  for (int i = 0; i < 4; ++i) WtB[i] = (u16*)take((size_t)F * F * 2); // W^T bf16 (B op)
  float*    f1 = (float*)take((size_t)Nn * 4);
  float*    f2 = (float*)take((size_t)Nn * 4);

  // one-time conversions / packing
  cvt_bf16_kernel<<<(Nn * F) / 256, 256, 0, stream>>>(input, xB, Nn * F);
  cvt_bf16_t_kernel<<<(F * F) / 256, 256, 0, stream>>>(W0, WtB[0], F, F);
  for (int l = 0; l < 3; ++l)
    cvt_bf16_t_kernel<<<(F * F) / 256, 256, 0, stream>>>(Wl[l], WtB[l + 1], F, F);
  pack_adj_kernel<<<(Nn * (Nn / 32)) / 256, 256, 0, stream>>>(adj, adjb);

  dim3 blk(256);
  dim3 grid(F / BN, Nn / BM);   // all GEMMs produce [4096 x 256]

  // h = input @ W0        (bf16 row-major out only)
  gemm_bf16_wmma<<<grid, blk, 0, stream>>>(xB, WtB[0], nullptr, hB, nullptr,
                                           Nn, F, F, /*act=*/0);

  for (int l = 0; l < 3; ++l) {
    // Wh = h @ W           (f32 for gemv + transposed bf16 as next B operand)
    gemm_bf16_wmma<<<grid, blk, 0, stream>>>(hB, WtB[l + 1], WhF, nullptr, WhBt,
                                             Nn, F, F, /*act=*/0);
    gemv_f1f2<<<Nn / 8, blk, 0, stream>>>(WhF, al[l], f1, f2);
    masked_softmax<<<Nn, blk, 0, stream>>>(f1, f2, adjb, att);
    // h' = ELU(att @ Wh)   (Bt = Wh^T, K = 4096)
    if (l == 2) {
      gemm_bf16_wmma<<<grid, blk, 0, stream>>>(att, WhBt, (float*)d_out, nullptr, nullptr,
                                               Nn, F, Nn, /*act=*/1);
    } else {
      gemm_bf16_wmma<<<grid, blk, 0, stream>>>(att, WhBt, nullptr, hB, nullptr,
                                               Nn, F, Nn, /*act=*/1);
    }
  }
}
